// Net_58033598104004
// MI455X (gfx1250) — matrix operand order repaired and checked by reading, not compiled, not run on hardware
//
#include <hip/hip_runtime.h>
#include <math.h>

// ---------------------------------------------------------------------------
// DGCNN-style Net on gfx1250: EdgeConv via v_wmma_f32_16x16x32_f16.
// Each node's 16 in-edges are contiguous (dst = e/16), so one wave owns one
// node: 16 edges = the M=16 rows of a WMMA tile. Weights are pre-packed into
// the exact B-fragment lane layout and held stationary in LDS. Bias+ReLU+BN
// and the masked segment-max are fused into the WMMA epilogue.
// Gather uses b128 global loads; A-staging uses b64 LDS stores.
// ---------------------------------------------------------------------------

typedef __attribute__((ext_vector_type(16))) _Float16 v16h;
typedef __attribute__((ext_vector_type(8)))  _Float16 v8h;
typedef __attribute__((ext_vector_type(4)))  _Float16 v4h;
typedef __attribute__((ext_vector_type(8)))  float    v8f;

#define KEDGE   16
#define BGRAPH  16
#define NNODE   2048
#define NORIG   (BGRAPH * NNODE)       // 32768
#define EPSBN   1e-5f

// ---------------------------------------------------------------------------
// Weight packing: fragment layout for B operand of WMMA_F32_16X16X32_F16.
// Tile (kt,nt): packed[tile*512 + lane*16 + j] = W[kt*32 + (lane>>4)*16 + j]
//                                                  [nt*16 + (lane&15)]
// Also folds BN (applied after ReLU): s = gamma*rsqrt(var+eps), t = beta-mean*s
// ---------------------------------------------------------------------------
struct LayerDesc {
    const float *W, *b, *beta, *gamma, *mean, *var;
    int Kreal, Kpad, Nout, woff, bstoff;
};
struct PackArgs {
    LayerDesc L[9];
    const float* pw[3];
};

__global__ __launch_bounds__(256) void pack_kernel(PackArgs pa, _Float16* wpack,
                                                   float* bst, float* wn) {
    int bid = blockIdx.x, tid = threadIdx.x;
    if (bid < 9) {
        LayerDesc d = pa.L[bid];
        int ntn   = d.Nout / 16;
        int total = (d.Kpad / 32) * ntn * 512;
        for (int idx = tid; idx < total; idx += blockDim.x) {
            int tile = idx >> 9;
            int rem  = idx & 511;
            int lane = rem >> 4;
            int j    = rem & 15;
            int kt = tile / ntn, nt = tile % ntn;
            int K  = kt * 32 + (lane >> 4) * 16 + j;
            int Nc = nt * 16 + (lane & 15);
            float v = (K < d.Kreal) ? d.W[K * d.Nout + Nc] : 0.f;
            wpack[d.woff + idx] = (_Float16)v;
        }
        for (int c = tid; c < d.Nout; c += blockDim.x) {
            float s = d.gamma[c] * rsqrtf(d.var[c] + EPSBN);
            float t = d.beta[c] - d.mean[c] * s;
            bst[d.bstoff + c]        = d.b[c];
            bst[d.bstoff + 64 + c]   = s;
            bst[d.bstoff + 128 + c]  = t;
        }
    } else {
        int p = bid - 9;
        const float* w = pa.pw[p];
        __shared__ float nrm;
        if (tid == 0) {
            float ss = 0.f;
            for (int c = 0; c < 64; ++c) ss += w[c] * w[c];
            nrm = sqrtf(ss);
        }
        __syncthreads();
        for (int c = tid; c < 64; c += blockDim.x) wn[p * 64 + c] = w[c] / nrm;
    }
}

__global__ void init_maps_kernel(int* node_map, int* cur2orig) {
    int v = blockIdx.x * blockDim.x + threadIdx.x;
    if (v < NORIG) { node_map[v] = v; cur2orig[v] = v; }
}

// ---------------------------------------------------------------------------
// One fused MLP layer on a 16-row tile: D = relu(A@W + b)*s + t.
// A in LDS, stride KD halves (this code's contiguous order: lane half hh reads k = 16*hh + 0..15).
// LAST: masked max over valid rows -> global h_out (isolated nodes -> 0).
// ---------------------------------------------------------------------------
template<int KD, int NO, bool LAST>
__device__ __forceinline__ void mlp_layer(const _Float16* __restrict__ A,
                                          const _Float16* __restrict__ sW, int wbase,
                                          const float* __restrict__ bst, int lane,
                                          _Float16* __restrict__ out,
                                          unsigned rowmask,
                                          float* __restrict__ hout, int node) {
    const int r  = lane & 15;   // A: row M   / CD: column N
    const int hh = lane >> 4;   // lane-half
#pragma unroll
    for (int nt = 0; nt < NO / 16; ++nt) {
        v8f acc = {0.f, 0.f, 0.f, 0.f, 0.f, 0.f, 0.f, 0.f};
#pragma unroll
        for (int kt = 0; kt < KD / 32; ++kt) {
            v16h af, bf;
            const _Float16* ap = A + r * KD + kt * 32 + hh * 16;
            ((v8h*)&af)[0] = *(const v8h*)(ap);        // K = kt*32 + hh*16 + 0..7
            ((v8h*)&af)[1] = *(const v8h*)(ap + 8);    // K = kt*32 + hh*16 + 8..15
            const _Float16* bp = sW + wbase + (kt * (NO / 16) + nt) * 512 + lane * 16;
            ((v8h*)&bf)[0] = *(const v8h*)(bp);
            ((v8h*)&bf)[1] = *(const v8h*)(bp + 8);
            acc = __builtin_amdgcn_wmma_f32_16x16x32_f16(
                false, af, false, bf, (short)0, acc, false, false);
        }
        int c = nt * 16 + r;
        float b = bst[c], s = bst[64 + c], t = bst[128 + c];
        if (!LAST) {
#pragma unroll
            for (int j = 0; j < 8; ++j) {
                int m = hh * 8 + j;
                float z = acc[j] + b;
                z = fmaxf(z, 0.f);
                out[m * NO + c] = (_Float16)(z * s + t);
            }
        } else {
            float mx = -INFINITY;
#pragma unroll
            for (int j = 0; j < 8; ++j) {
                int m = hh * 8 + j;
                float z = acc[j] + b;
                z = fmaxf(z, 0.f);
                z = z * s + t;
                if (rowmask & (1u << m)) mx = fmaxf(mx, z);
            }
            mx = fmaxf(mx, __shfl_xor(mx, 16, 32));
            if (hh == 0) hout[node * 64 + c] = (mx == -INFINITY) ? 0.f : mx;
        }
    }
}

// ---------------------------------------------------------------------------
// EdgeConv: block = 4 waves, one node per wave. Weights stationary in LDS.
// FIRST: features = [pos(3), x(3)], 2C=12 padded to K=32. Else 2C=128.
// ---------------------------------------------------------------------------
template<bool FIRST>
__global__ __launch_bounds__(128) void edgeconv_kernel(
    int nodes, const float* __restrict__ hin, const float* __restrict__ xin,
    const int* __restrict__ esrc, const int* __restrict__ cur2orig,
    const int* __restrict__ node_map, const _Float16* __restrict__ wpack,
    const float* __restrict__ bst, float* __restrict__ hout) {
    constexpr int WTOT = FIRST ? 7168 : 16384;     // packed halves this conv
    __shared__ _Float16 sW[16384];                 // 32 KB weights
    __shared__ _Float16 sA[4][2048];               // per-wave A scratch (16x128)
    __shared__ _Float16 sO[4][1024];               // per-wave inter-layer (16x64)

    int tid = threadIdx.x;
    for (int i = tid; i < WTOT / 8; i += blockDim.x)
        ((v8h*)sW)[i] = ((const v8h*)wpack)[i];
    __syncthreads();

    int w = tid >> 5, lane = tid & 31;
    int node = blockIdx.x * 4 + w;
    if (node >= nodes) return;                     // wave-uniform

    int orig  = FIRST ? node : cur2orig[node];
    int ebase = orig * KEDGE;

    unsigned rowmask;
    if (FIRST) {
        rowmask = 0xFFFFu;
    } else {
        int valid = 0;
        if (lane < 16) valid = (node_map[esrc[ebase + lane]] >= 0) ? 1 : 0;
        rowmask = (unsigned)__ballot(valid) & 0xFFFFu;
    }

    _Float16* A0 = sA[w];
    if (FIRST) {
        if (lane < 16) {
            int r  = lane;
            int s0 = esrc[ebase + r];
            float xi[6], xs[6];
#pragma unroll
            for (int c = 0; c < 3; ++c) {
                xi[c]     = hin[node * 3 + c];   // pos
                xi[3 + c] = xin[node * 3 + c];   // x
                xs[c]     = hin[s0 * 3 + c];
                xs[3 + c] = xin[s0 * 3 + c];
            }
            _Float16* row = A0 + r * 32;
#pragma unroll
            for (int c = 0; c < 6; ++c) {
                row[c]     = (_Float16)xi[c];
                row[6 + c] = (_Float16)(xs[c] - xi[c]);
            }
#pragma unroll
            for (int c = 12; c < 32; ++c) row[c] = (_Float16)0.f;
        }
    } else {
        int r = lane >> 1, hs = lane & 1;          // 2 lanes per row
        int s0 = esrc[ebase + r];
        int sm = node_map[s0];
        if (sm < 0) sm = 0;
        const float4* xi4 = (const float4*)(hin + node * 64 + hs * 32);
        const float4* xs4 = (const float4*)(hin + sm * 64 + hs * 32);
        _Float16* rb = A0 + r * 128 + hs * 32;
#pragma unroll
        for (int q = 0; q < 8; ++q) {
            float4 a = xi4[q];                     // b128 gather
            float4 b = xs4[q];
            v4h ha = { (_Float16)a.x, (_Float16)a.y,
                       (_Float16)a.z, (_Float16)a.w };
            v4h hd = { (_Float16)(b.x - a.x), (_Float16)(b.y - a.y),
                       (_Float16)(b.z - a.z), (_Float16)(b.w - a.w) };
            *(v4h*)(rb + q * 4)      = ha;         // x_i        (b64 store)
            *(v4h*)(rb + 64 + q * 4) = hd;         // x_src - x_i
        }
    }

    if (FIRST) {
        mlp_layer<32, 32, false>(A0,    sW, 0,     bst + 0,   lane, sO[w], 0, nullptr, 0);
        mlp_layer<32, 64, false>(sO[w], sW, 1024,  bst + 192, lane, sA[w], 0, nullptr, 0);
        mlp_layer<64, 64, true >(sA[w], sW, 3072,  bst + 384, lane, nullptr, rowmask, hout, node);
    } else {
        mlp_layer<128, 64, false>(A0,    sW, 0,     bst + 0,   lane, sO[w], 0, nullptr, 0);
        mlp_layer<64,  64, false>(sO[w], sW, 8192,  bst + 192, lane, sA[w], 0, nullptr, 0);
        mlp_layer<64,  64, true >(sA[w], sW, 12288, bst + 384, lane, nullptr, rowmask, hout, node);
    }
}

// ---------------------------------------------------------------------------
// TopK pool: per-graph bitonic sort of keys (score desc, index asc ties).
// Writes scaled features, old->new map for this stage, new cur->orig map.
// ---------------------------------------------------------------------------
__global__ __launch_bounds__(256) void pool_kernel(
    int n, int k, const float* __restrict__ hin, const float* __restrict__ wn,
    float* __restrict__ hout, const int* __restrict__ cur2orig,
    int* __restrict__ new_cur2orig, int* __restrict__ stage_old2new) {
    __shared__ unsigned long long kk[2048];
    __shared__ float sc[2048];
    int g = blockIdx.x, tid = threadIdx.x;

    for (int i = tid; i < n; i += 256) {
        float acc = 0.f;
        const float4* hp4 = (const float4*)(hin + (size_t)(g * n + i) * 64);
        const float4* wn4 = (const float4*)wn;
#pragma unroll
        for (int c = 0; c < 16; ++c) {
            float4 h4 = hp4[c], w4 = wn4[c];
            acc += h4.x * w4.x + h4.y * w4.y + h4.z * w4.z + h4.w * w4.w;
        }
        float sv = tanhf(acc);
        sc[i] = sv;
        unsigned u = __float_as_uint(sv);
        u = (u & 0x80000000u) ? ~u : (u | 0x80000000u);   // ascending-sortable
        kk[i] = ((unsigned long long)(0xFFFFFFFFu - u) << 32) | (unsigned)i;
        stage_old2new[g * n + i] = -1;
    }
    __syncthreads();

    for (int size = 2; size <= n; size <<= 1)
        for (int stride = size >> 1; stride > 0; stride >>= 1) {
            for (int i = tid; i < n; i += 256) {
                int j = i ^ stride;
                if (j > i) {
                    bool up = ((i & size) == 0);
                    unsigned long long a = kk[i], b = kk[j];
                    if ((a > b) == up) { kk[i] = b; kk[j] = a; }
                }
            }
            __syncthreads();
        }

    for (int j = tid; j < k; j += 256) {
        int old = (int)(kk[j] & 0xFFFFFFFFull);
        int nw  = g * k + j;
        stage_old2new[g * n + old] = nw;
        new_cur2orig[nw] = cur2orig[g * n + old];
    }
    __syncthreads();
    for (int idx = tid; idx < k * 16; idx += 256) {        // float4 granules
        int j = idx >> 4, c = idx & 15;
        int old = (int)(kk[j] & 0xFFFFFFFFull);
        float4 v = ((const float4*)(hin + (size_t)(g * n + old) * 64))[c];
        float s = sc[old];
        v.x *= s; v.y *= s; v.z *= s; v.w *= s;
        ((float4*)(hout + (size_t)(g * k + j) * 64))[c] = v;
    }
}

__global__ void map_update_kernel(int bk, int* node_map, const int* stage_old2new,
                                  int* cur2orig, const int* new_cur2orig) {
    int v = blockIdx.x * blockDim.x + threadIdx.x;
    if (v < NORIG) {
        int m = node_map[v];
        node_map[v] = (m >= 0) ? stage_old2new[m] : -1;
    }
    if (v < bk) cur2orig[v] = new_cur2orig[v];
}

__global__ __launch_bounds__(64) void readout_kernel(int n, const float* __restrict__ h,
                                                     float* __restrict__ xo) {
    int g = blockIdx.x, c = threadIdx.x;
    float mx = -INFINITY, sm = 0.f;
    for (int i = 0; i < n; ++i) {
        float v = h[(size_t)(g * n + i) * 64 + c];
        mx = fmaxf(mx, v);
        sm += v;
    }
    xo[g * 128 + c]      = mx;
    xo[g * 128 + 64 + c] = sm / (float)n;
}

// ---------------------------------------------------------------------------
// Head: z = x1+x2+x3 -> 128->64->32->40 -> log_softmax. B=16, trivially small.
// ---------------------------------------------------------------------------
__global__ __launch_bounds__(256) void head_kernel(
    const float* x1, const float* x2, const float* x3,
    const float* W1, const float* b1, const float* W2, const float* b2,
    const float* W3, const float* b3, float* out) {
    __shared__ float Z[16 * 128], Z2[16 * 64], Z3[16 * 32], LG[16 * 40];
    int tid = threadIdx.x;
    for (int i = tid; i < 16 * 128; i += 256) Z[i] = x1[i] + x2[i] + x3[i];
    __syncthreads();
    for (int t = tid; t < 16 * 64; t += 256) {
        int g = t >> 6, c = t & 63;
        float acc = b1[c];
        for (int i = 0; i < 128; ++i) acc += Z[g * 128 + i] * W1[i * 64 + c];
        Z2[t] = fmaxf(acc, 0.f);
    }
    __syncthreads();
    for (int t = tid; t < 16 * 32; t += 256) {
        int g = t >> 5, c = t & 31;
        float acc = b2[c];
        for (int i = 0; i < 64; ++i) acc += Z2[g * 64 + i] * W2[i * 32 + c];
        Z3[t] = fmaxf(acc, 0.f);
    }
    __syncthreads();
    for (int t = tid; t < 16 * 40; t += 256) {
        int g = t / 40, c = t % 40;
        float acc = b3[c];
        for (int i = 0; i < 32; ++i) acc += Z3[g * 32 + i] * W3[i * 40 + c];
        LG[t] = acc;
    }
    __syncthreads();
    if (tid < 16) {
        int g = tid;
        float mx = -INFINITY;
        for (int c = 0; c < 40; ++c) mx = fmaxf(mx, LG[g * 40 + c]);
        float se = 0.f;
        for (int c = 0; c < 40; ++c) se += expf(LG[g * 40 + c] - mx);
        float lse = mx + logf(se);
        for (int c = 0; c < 40; ++c) out[g * 40 + c] = LG[g * 40 + c] - lse;
    }
}

// ---------------------------------------------------------------------------
// Launch. d_in order (jax dict flatten, keys sorted; 'W' < 'b' < 'beta' < ...):
// 0:x 1:pos 2:edge_src 3:edge_dst
// 4..21 conv1 (3 layers x [W,b,beta,gamma,mean,var]), 22..39 conv2, 40..57 conv3
// 58,59 lin1 W,b  60,61 lin2  62,63 lin3  64,65,66 pool1_w/2/3
// ---------------------------------------------------------------------------
extern "C" void kernel_launch(void* const* d_in, const int* in_sizes, int n_in,
                              void* d_out, int out_size, void* d_ws, size_t ws_size,
                              hipStream_t stream) {
    (void)in_sizes; (void)n_in; (void)out_size; (void)ws_size;
    const float* x   = (const float*)d_in[0];
    const float* pos = (const float*)d_in[1];
    const int*   esrc = (const int*)d_in[2];
    auto leaf = [&](int i) { return (const float*)d_in[i]; };

    PackArgs pa;
    const int convW0[3] = {4, 22, 40};
    const int Kreal[9] = {12, 32, 64, 128, 64, 64, 128, 64, 64};
    const int Kpad [9] = {32, 32, 64, 128, 64, 64, 128, 64, 64};
    const int Nout [9] = {32, 64, 64, 64, 64, 64, 64, 64, 64};
    const int woff [9] = {0, 1024, 3072, 8192, 16384, 20480, 24576, 32768, 36864};
    for (int cv = 0; cv < 3; ++cv)
        for (int L = 0; L < 3; ++L) {
            int li = cv * 3 + L, base = convW0[cv] + L * 6;
            pa.L[li].W = leaf(base + 0); pa.L[li].b = leaf(base + 1);
            pa.L[li].beta = leaf(base + 2); pa.L[li].gamma = leaf(base + 3);
            pa.L[li].mean = leaf(base + 4); pa.L[li].var = leaf(base + 5);
            pa.L[li].Kreal = Kreal[li]; pa.L[li].Kpad = Kpad[li];
            pa.L[li].Nout = Nout[li]; pa.L[li].woff = woff[li];
            pa.L[li].bstoff = li * 192;
        }
    pa.pw[0] = leaf(64); pa.pw[1] = leaf(65); pa.pw[2] = leaf(66);

    char* ws = (char*)d_ws;
    _Float16* wpack  = (_Float16*)(ws + 0);              // 80 KB used
    float* bst       = (float*)(ws + (128ull << 10));
    float* wn        = (float*)(ws + (144ull << 10));
    int* node_map    = (int*)(ws + (148ull << 10));
    int* cur2orig    = (int*)(ws + (276ull << 10));
    int* new_c2o     = (int*)(ws + (404ull << 10));
    int* st_o2n      = (int*)(ws + (468ull << 10));
    float* xr1       = (float*)(ws + (596ull << 10));
    float* xr2       = (float*)(ws + (604ull << 10));
    float* xr3       = (float*)(ws + (612ull << 10));
    float* h0        = (float*)(ws + (1ull  << 20));     // 32768 x 64
    float* h1        = (float*)(ws + (9ull  << 20));     // 16384 x 64
    float* h1c       = (float*)(ws + (13ull << 20));
    float* h2        = (float*)(ws + (17ull << 20));     // 8192 x 64
    float* h2c       = (float*)(ws + (19ull << 20));
    float* h3        = (float*)(ws + (21ull << 20));     // 4096 x 64

    pack_kernel<<<12, 256, 0, stream>>>(pa, wpack, bst, wn);
    init_maps_kernel<<<128, 256, 0, stream>>>(node_map, cur2orig);

    // Stage 1
    edgeconv_kernel<true><<<32768 / 4, 128, 0, stream>>>(
        32768, pos, x, esrc, cur2orig, node_map, wpack + 0, bst + 0, h0);
    pool_kernel<<<16, 256, 0, stream>>>(2048, 1024, h0, wn + 0, h1,
                                        cur2orig, new_c2o, st_o2n);
    map_update_kernel<<<128, 256, 0, stream>>>(16 * 1024, node_map, st_o2n,
                                               cur2orig, new_c2o);
    readout_kernel<<<16, 64, 0, stream>>>(1024, h1, xr1);

    // Stage 2
    edgeconv_kernel<false><<<16384 / 4, 128, 0, stream>>>(
        16384, h1, nullptr, esrc, cur2orig, node_map, wpack + 8192, bst + 576, h1c);
    pool_kernel<<<16, 256, 0, stream>>>(1024, 512, h1c, wn + 64, h2,
                                        cur2orig, new_c2o, st_o2n);
    map_update_kernel<<<128, 256, 0, stream>>>(16 * 512, node_map, st_o2n,
                                               cur2orig, new_c2o);
    readout_kernel<<<16, 64, 0, stream>>>(512, h2, xr2);

    // Stage 3
    edgeconv_kernel<false><<<8192 / 4, 128, 0, stream>>>(
        8192, h2, nullptr, esrc, cur2orig, node_map, wpack + 24576, bst + 1152, h2c);
    pool_kernel<<<16, 256, 0, stream>>>(512, 256, h2c, wn + 128, h3,
                                        cur2orig, new_c2o, st_o2n);
    map_update_kernel<<<128, 256, 0, stream>>>(16 * 256, node_map, st_o2n,
                                               cur2orig, new_c2o);
    readout_kernel<<<16, 64, 0, stream>>>(256, h3, xr3);

    head_kernel<<<1, 256, 0, stream>>>(xr1, xr2, xr3, leaf(58), leaf(59),
                                       leaf(60), leaf(61), leaf(62), leaf(63),
                                       (float*)d_out);
}
